// Head_9199819948660
// MI455X (gfx1250) — matrix operand order repaired and checked by reading, not compiled, not run on hardware
//
#include <hip/hip_runtime.h>
#include <hip/hip_bf16.h>

typedef __attribute__((ext_vector_type(16))) _Float16 v16h;
typedef __attribute__((ext_vector_type(8)))  _Float16 v8h;
typedef __attribute__((ext_vector_type(2)))  _Float16 v2h;
typedef __attribute__((ext_vector_type(8)))  float    v8f;

constexpr int Bb = 8;
constexpr int Tt = 2048;
constexpr int Dd = 1024;
constexpr int Hh = 64;

// A-fragment k index for element j (16-bit A 16x32, lane half hf):
//   k = 16*hf + j  (contiguous, same order as B)
// B-fragment k index for element j: k = 16*hf + j  (16 contiguous halfs)

// ---------------------------------------------------------------------------
// Phase 1: Q = xWq, K = xWk, V = xWv  (fp32 in -> f16 out; V stored transposed)
// 128 threads (4 waves); block computes a 16-row tile; wave w owns cols 16w..
// x staged in LDS in A-fragment order; W staged transposed (B-fragment order).
// ---------------------------------------------------------------------------
__global__ __launch_bounds__(128) void qkv_kernel(
    const float* __restrict__ x,  const float* __restrict__ Wq,
    const float* __restrict__ Wk, const float* __restrict__ Wv,
    _Float16* __restrict__ Qh, _Float16* __restrict__ Kh,
    _Float16* __restrict__ Vt)
{
    __shared__ __align__(32) _Float16 xsf[32][16];   // x chunk, A-fragment order
    __shared__ __align__(32) _Float16 wqT[64][32];   // W chunk, transposed [n][k]
    __shared__ __align__(32) _Float16 wkT[64][32];
    __shared__ __align__(32) _Float16 wvT[64][32];

    const int tid  = threadIdx.x;
    const int wave = tid >> 5;
    const int lane = tid & 31;
    const int b    = blockIdx.x >> 7;            // T/16 = 128 row tiles
    const int rt   = blockIdx.x & 127;
    const int row0 = rt << 4;

    const int n_b   = (wave << 4) + (lane & 15); // B-fragment column
    const int bbase = (lane >> 4) << 4;          // B-fragment k base (0/16)

    v8f accq = {}; v8f acck = {}; v8f accv = {};

    for (int kk = 0; kk < Dd; kk += 32) {
        // ---- stage x: 256 half2 pairs in A-fragment order ----
        for (int e = tid; e < 256; e += 128) {
            int lane_i = e >> 3;                 // fragment lane
            int j0     = (e & 7) << 1;           // even element index
            int m      = lane_i & 15;
            int hf     = lane_i >> 4;
            int k0     = hf * 16 + j0;
            const float* src = x + ((size_t)b * Tt + row0 + m) * Dd + kk + k0;
            v2h t = { (_Float16)src[0], (_Float16)src[1] };
            *(v2h*)&xsf[lane_i][j0] = t;
        }
        // ---- stage weights transposed: wT[n][k] = W[kk+k][n] ----
        for (int e = tid; e < 1024; e += 128) {
            int n  = e & 63;
            int k  = (e >> 6) << 1;              // 0,2,..,30
            int gi = (kk + k) * Hh + n;
            v2h q = { (_Float16)Wq[gi], (_Float16)Wq[gi + Hh] };
            v2h kx = { (_Float16)Wk[gi], (_Float16)Wk[gi + Hh] };
            v2h vx = { (_Float16)Wv[gi], (_Float16)Wv[gi + Hh] };
            *(v2h*)&wqT[n][k] = q;
            *(v2h*)&wkT[n][k] = kx;
            *(v2h*)&wvT[n][k] = vx;
        }
        __syncthreads();

        v16h a  = *(const v16h*)&xsf[lane][0];
        v16h bq = *(const v16h*)&wqT[n_b][bbase];
        v16h bk = *(const v16h*)&wkT[n_b][bbase];
        v16h bv = *(const v16h*)&wvT[n_b][bbase];

        accq = __builtin_amdgcn_wmma_f32_16x16x32_f16(false, a, false, bq, (short)0, accq, false, false);
        acck = __builtin_amdgcn_wmma_f32_16x16x32_f16(false, a, false, bk, (short)0, acck, false, false);
        accv = __builtin_amdgcn_wmma_f32_16x16x32_f16(false, a, false, bv, (short)0, accv, false, false);
        __syncthreads();
    }

    // C-layout: VGPR r holds row m = r + 8*(lane>=16), column n = n_b.
    const int hl = lane >> 4;
    #pragma unroll
    for (int r = 0; r < 8; ++r) {
        int m = r + (hl << 3);
        size_t row = (size_t)b * Tt + row0 + m;
        Qh[row * Hh + n_b] = (_Float16)accq[r];
        Kh[row * Hh + n_b] = (_Float16)acck[r];
    }
    // V transposed: lane's 8 rows are 8 consecutive keys -> one b128 store.
    {
        v8h vv;
        #pragma unroll
        for (int r = 0; r < 8; ++r) vv[r] = (_Float16)accv[r];
        *(v8h*)(Vt + ((size_t)b * Hh + n_b) * Tt + row0 + (hl << 3)) = vv;
    }
}

// ---------------------------------------------------------------------------
// Phase 2: flash attention, causal + key padding mask.
// One wave per 16-row query tile; 32-key tiles so P (16x32 f16) is a full
// K=32 A-operand. P bounced through LDS in A-fragment order.
// ---------------------------------------------------------------------------
__global__ __launch_bounds__(32) void attn_kernel(
    const _Float16* __restrict__ Qh, const _Float16* __restrict__ Kh,
    const _Float16* __restrict__ Vt, const unsigned char* __restrict__ kpm,
    float* __restrict__ out)
{
    __shared__ __align__(32) _Float16 Pf[32][16];   // P in A-fragment order

    const int lane  = threadIdx.x;
    const int b     = blockIdx.x >> 7;
    const int qt    = blockIdx.x & 127;
    const int q0    = qt << 4;
    const int hl    = lane >> 4;
    const int ln    = lane & 15;
    const int abase = hl << 3;                  // unused; A-fragments also use k base 0/16
    const int bbase = hl << 4;                  // B-fragment k base (0/16)

    // Q A-fragments: two aligned v8h loads per 32-chunk, concatenated.
    v16h qf[2];
    {
        const _Float16* qrow = Qh + ((size_t)b * Tt + q0 + ln) * Hh;
        #pragma unroll
        for (int c = 0; c < 2; ++c) {
            v8h lo = *(const v8h*)(qrow + c * 32 + bbase);
            v8h hi = *(const v8h*)(qrow + c * 32 + bbase + 8);
            qf[c] = __builtin_shufflevector(lo, hi, 0, 1, 2, 3, 4, 5, 6, 7,
                                            8, 9, 10, 11, 12, 13, 14, 15);
        }
    }

    float mrow[8], lrow[8];
    #pragma unroll
    for (int r = 0; r < 8; ++r) { mrow[r] = -1e30f; lrow[r] = 0.0f; }
    v8f o[4] = {};

    const float scale = 0.125f;                 // 1/sqrt(64)

    for (int kb = 0; kb < q0 + 16; kb += 32) {
        if (kb + 32 < q0 + 16) {                // prefetch next K/V tiles
            __builtin_prefetch(Kh + ((size_t)b * Tt + kb + 32 + ln) * Hh, 0, 3);
            __builtin_prefetch(Vt + ((size_t)b * Hh + ln) * Tt + kb + 32, 0, 3);
        }

        // ---- S = Q @ K^T: B-fragments are contiguous 32B in K rows ----
        v8f s[2];
        #pragma unroll
        for (int nt = 0; nt < 2; ++nt) {
            const _Float16* krow = Kh + ((size_t)b * Tt + kb + nt * 16 + ln) * Hh;
            v16h kf0 = *(const v16h*)(krow + bbase);
            v16h kf1 = *(const v16h*)(krow + 32 + bbase);
            v8f acc = {};
            acc = __builtin_amdgcn_wmma_f32_16x16x32_f16(false, qf[0], false, kf0, (short)0, acc, false, false);
            acc = __builtin_amdgcn_wmma_f32_16x16x32_f16(false, qf[1], false, kf1, (short)0, acc, false, false);
            s[nt] = acc;
        }

        // ---- scale + causal/padding mask ----
        float p[2][8];
        #pragma unroll
        for (int nt = 0; nt < 2; ++nt) {
            int key = kb + nt * 16 + ln;
            bool padded = kpm[(size_t)b * Tt + key] != 0;
            #pragma unroll
            for (int r = 0; r < 8; ++r) {
                int qrow = q0 + r + (hl << 3);
                float sv = s[nt][r] * scale;
                if (key > qrow || padded) sv = -1e30f;
                p[nt][r] = sv;
            }
        }

        // ---- online softmax (row stats via 16-lane shuffle reductions) ----
        float alpha[8];
        #pragma unroll
        for (int r = 0; r < 8; ++r) {
            float t = fmaxf(p[0][r], p[1][r]);
            #pragma unroll
            for (int off = 8; off >= 1; off >>= 1)
                t = fmaxf(t, __shfl_xor(t, off, 32));
            float mn = fmaxf(mrow[r], t);
            alpha[r] = __expf(mrow[r] - mn);
            mrow[r]  = mn;
            p[0][r] = __expf(p[0][r] - mn);
            p[1][r] = __expf(p[1][r] - mn);
            float ts = p[0][r] + p[1][r];
            #pragma unroll
            for (int off = 8; off >= 1; off >>= 1)
                ts += __shfl_xor(ts, off, 32);
            lrow[r] = lrow[r] * alpha[r] + ts;
        }
        #pragma unroll
        for (int t = 0; t < 4; ++t)
            #pragma unroll
            for (int r = 0; r < 8; ++r)
                o[t][r] *= alpha[r];

        // ---- scatter P (C-layout) into LDS in A-fragment order ----
        #pragma unroll
        for (int nt = 0; nt < 2; ++nt) {
            int kc = nt * 16 + ln;                    // this lane's score column
            int hf = kc >> 4;
            int j  = kc & 15;
            #pragma unroll
            for (int r = 0; r < 8; ++r) {
                int m = r + (hl << 3);
                Pf[m + (hf << 4)][j] = (_Float16)p[nt][r];
            }
        }
        __syncthreads();

        v16h pf = *(const v16h*)&Pf[lane][0];         // one aligned fragment load

        // ---- O += P @ V : V^T rows give contiguous 32B B-fragments ----
        #pragma unroll
        for (int t = 0; t < 4; ++t) {
            const _Float16* vrow =
                Vt + ((size_t)b * Hh + t * 16 + ln) * Tt + kb + bbase;
            v16h vf = *(const v16h*)vrow;
            o[t] = __builtin_amdgcn_wmma_f32_16x16x32_f16(false, pf, false, vf, (short)0, o[t], false, false);
        }
        __syncthreads();
    }

    // ---- epilogue: out = O / l ----
    #pragma unroll
    for (int r = 0; r < 8; ++r) {
        float inv = 1.0f / lrow[r];
        int m = r + (hl << 3);
        size_t orow = ((size_t)b * Tt + q0 + m) * Hh;
        #pragma unroll
        for (int t = 0; t < 4; ++t)
            out[orow + t * 16 + ln] = o[t][r] * inv;
    }
}

extern "C" void kernel_launch(void* const* d_in, const int* in_sizes, int n_in,
                              void* d_out, int out_size, void* d_ws, size_t ws_size,
                              hipStream_t stream) {
    const float* x  = (const float*)d_in[0];
    const float* Wq = (const float*)d_in[1];
    const float* Wk = (const float*)d_in[2];
    const float* Wv = (const float*)d_in[3];
    const unsigned char* kpm = (const unsigned char*)d_in[4];
    float* out = (float*)d_out;

    // Workspace: Q (f16), K (f16), V^T (f16) => 6 MB total.
    const size_t nQ = (size_t)Bb * Tt * Hh;
    _Float16* Qh = (_Float16*)d_ws;
    _Float16* Kh = Qh + nQ;
    _Float16* Vt = Kh + nQ;

    qkv_kernel<<<dim3(Bb * (Tt / 16)), dim3(128), 0, stream>>>(x, Wq, Wk, Wv, Qh, Kh, Vt);
    attn_kernel<<<dim3(Bb * (Tt / 16)), dim3(32), 0, stream>>>(Qh, Kh, Vt, kpm, out);
}
